// TransformerEncoderLayer_19851338842358
// MI455X (gfx1250) — matrix operand order repaired and checked by reading, not compiled, not run on hardware
//
#include <hip/hip_runtime.h>
#include <hip/hip_bf16.h>

typedef __bf16 bf16_t;
typedef __bf16 v16bf __attribute__((ext_vector_type(16)));
typedef float  v8f   __attribute__((ext_vector_type(8)));
typedef unsigned int u32x4 __attribute__((ext_vector_type(4)));
typedef int i32x4 __attribute__((ext_vector_type(4)));
typedef int i32x8 __attribute__((ext_vector_type(8)));

#define LQ_   2048
#define BZ_   8
#define DM_   1024
#define NH_   16
#define NK_   4
#define FF_   2048
#define HD_   64
#define MROWS (LQ_*BZ_)

#if defined(__has_builtin)
#if __has_builtin(__builtin_amdgcn_tensor_load_to_lds) && __has_builtin(__builtin_amdgcn_s_wait_tensorcnt)
#define USE_TDM 1
#endif
#endif
#ifndef USE_TDM
#define USE_TDM 0
#endif

union FragAB { v16bf v; unsigned int u[8]; };

// ---------------- elementwise helpers ----------------

__global__ void k_cvt_tr_bf16(const float* __restrict__ in,  // [K,N] row-major
                              bf16_t* __restrict__ out,      // [N,K] row-major
                              int K, int N) {
  long i = (long)blockIdx.x * blockDim.x + threadIdx.x;
  if (i < (long)K * N) {
    int k = (int)(i / N);
    int n = (int)(i - (long)k * N);
    out[(size_t)n * K + k] = (bf16_t)in[i];
  }
}

__global__ void k_prep(const float* __restrict__ src, const float* __restrict__ pos,
                       bf16_t* __restrict__ qb, bf16_t* __restrict__ sb, long n) {
  long i = (long)blockIdx.x * blockDim.x + threadIdx.x;
  if (i < n) {
    float s = src[i];
    qb[i] = (bf16_t)(s + pos[i]);
    sb[i] = (bf16_t)s;
  }
}

// ---------------- TDM tile load ----------------
// Loads a [rows<=128] x 64B tile (32 bf16 per row) from global (row stride Kd*2 bytes)
// into LDS with a 4-byte pad after every 64B row -> LDS row stride 68B (34 bf16).
// D# per CDNA5 ISA sec.8: data_size=3 (8B units), tile_dim0=8 units,
// tile_dim1=128, pad_enable=1, pad_interval=3 (16 DWORDs), pad_amount=0 (1 DWORD).
// tensor_dim1 = rows_avail gives hardware zero-fill for OOB rows.

#if USE_TDM
__device__ __forceinline__ void tdm_load_tile(const bf16_t* gbase, size_t row0,
                                              int rows_avail, int Kd, int k0,
                                              unsigned lds_addr) {
  unsigned long long ga = (unsigned long long)(const void*)gbase +
                          ((row0 * (size_t)Kd + (size_t)k0) << 1);
  u32x4 g0;
  g0.x = 1u;                                        // count=1
  g0.y = lds_addr;                                  // LDS byte address
  g0.z = (unsigned)(ga & 0xFFFFFFFFull);            // global_addr[31:0]
  g0.w = (unsigned)((ga >> 32) & 0x1FFFFFFull) | (2u << 30);  // addr[56:32] | type=2

  unsigned td1 = (unsigned)rows_avail;
  unsigned stride = (unsigned)(Kd >> 2);            // row stride in 8B units
  i32x8 g1;
  g1[0] = (int)((3u << 16) | (1u << 20) | (3u << 22));  // data_size=3,pad_en,pad_int=3
  g1[1] = (int)(8u << 16);                              // tensor_dim0[15:0]=8
  g1[2] = (int)((td1 & 0xFFFFu) << 16);                 // tensor_dim1[15:0]
  g1[3] = (int)(((td1 >> 16) & 0xFFFFu) | (8u << 16));  // tile_dim0=8 (64B)
  g1[4] = 128;                                          // tile_dim1=128, tile_dim2=0
  g1[5] = (int)stride;                                  // tensor_dim0_stride[31:0]
  g1[6] = 0;
  g1[7] = 0;

  i32x4 z4 = {0, 0, 0, 0};
#if __clang_major__ >= 23
  i32x8 z8 = {0, 0, 0, 0, 0, 0, 0, 0};
  __builtin_amdgcn_tensor_load_to_lds(g0, g1, z4, z4, z8, 0);
#else
  __builtin_amdgcn_tensor_load_to_lds(g0, g1, z4, z4, 0);
#endif
}
#endif

// ---------------- WMMA bf16 GEMM: C[M,N] = A[M,K] * Bt[N,K]^T + bias ----------------
// Block = 256 threads = 8 waves, tile 128x128, K-step 32, double-buffered LDS.
// Both A and Bt tiles are [128 rows][32 k] contiguous blocks -> identical staging.

template<bool RELU, bool OUT_BF16>
__global__ __launch_bounds__(256)
void k_gemm_bf16(const bf16_t* __restrict__ A, const bf16_t* __restrict__ Bt,
                 const float* __restrict__ bias, float* __restrict__ Cf,
                 bf16_t* __restrict__ Cb, int Mm, int Nn, int Kd) {
  __shared__ bf16_t Al[2][128 * 34];
  __shared__ bf16_t Bl[2][128 * 34];

  const int tid  = threadIdx.x;
  const int lane = tid & 31;
  const int wid  = tid >> 5;
  const int wm   = wid >> 2;
  const int wn   = wid & 3;
  const int m0   = blockIdx.y * 128;
  const int n0   = blockIdx.x * 128;
  const int half = lane >> 4;
  const int l16  = lane & 15;

  int rowsB = Nn - n0;
  if (rowsB > 128) rowsB = 128;

  v8f acc[4][2];
  for (int mi = 0; mi < 4; ++mi)
    for (int ni = 0; ni < 2; ++ni)
      for (int r = 0; r < 8; ++r) acc[mi][ni][r] = 0.0f;

  const int nIter = Kd >> 5;

#if USE_TDM
  const unsigned ldsA0 = (unsigned)(unsigned long long)(const void*)&Al[0][0];
  const unsigned ldsA1 = (unsigned)(unsigned long long)(const void*)&Al[1][0];
  const unsigned ldsB0 = (unsigned)(unsigned long long)(const void*)&Bl[0][0];
  const unsigned ldsB1 = (unsigned)(unsigned long long)(const void*)&Bl[1][0];
  if (wid == 0) {
    tdm_load_tile(A,  (size_t)m0, 128,   Kd, 0, ldsA0);
    tdm_load_tile(Bt, (size_t)n0, rowsB, Kd, 0, ldsB0);
  }
#endif

  for (int it = 0; it < nIter; ++it) {
    const int buf = it & 1;
    const int k0 = it << 5;

#if USE_TDM
    if (wid == 0) __builtin_amdgcn_s_wait_tensorcnt(0);   // tile `it` arrived
    __syncthreads();
    if (wid == 0 && it + 1 < nIter) {                      // prefetch tile it+1
      tdm_load_tile(A,  (size_t)m0, 128,   Kd, k0 + 32, buf ? ldsA0 : ldsA1);
      tdm_load_tile(Bt, (size_t)n0, rowsB, Kd, k0 + 32, buf ? ldsB0 : ldsB1);
    }
#else
    // fallback staging: contiguous 16B chunks, hardware-free zero fill emulated
#pragma unroll
    for (int itc = 0; itc < 2; ++itc) {
      int id = itc * 256 + tid;
      int r  = id >> 2;
      int c  = (id & 3) << 3;
      uint4 va = *(const uint4*)(A + (size_t)(m0 + r) * Kd + k0 + c);
      unsigned int* da = (unsigned int*)(&Al[buf][0] + r * 34 + c);
      da[0] = va.x; da[1] = va.y; da[2] = va.z; da[3] = va.w;
      uint4 vb;
      if (r < rowsB) vb = *(const uint4*)(Bt + (size_t)(n0 + r) * Kd + k0 + c);
      else { vb.x = 0u; vb.y = 0u; vb.z = 0u; vb.w = 0u; }
      unsigned int* db = (unsigned int*)(&Bl[buf][0] + r * 34 + c);
      db[0] = vb.x; db[1] = vb.y; db[2] = vb.z; db[3] = vb.w;
    }
    __syncthreads();
#endif

    // ---- fragments + WMMA ----
    FragAB a[4], b[2];
#pragma unroll
    for (int mi = 0; mi < 4; ++mi) {
      int row = wm * 64 + mi * 16 + l16;
#pragma unroll
      for (int r = 0; r < 8; ++r) {
        // A 16x32 bf16: K = (r/4)*16 + half*8 + (r%4)*2
        int kb = ((r >> 2) << 4) + (half << 3) + ((r & 3) << 1);
        a[mi].u[r] = *(const unsigned int*)(&Al[buf][0] + row * 34 + kb);
      }
    }
#pragma unroll
    for (int ni = 0; ni < 2; ++ni) {
      int col = wn * 32 + ni * 16 + l16;
      int ks  = half << 4;
#pragma unroll
      for (int e = 0; e < 8; ++e)
        b[ni].u[e] = *(const unsigned int*)(&Bl[buf][0] + col * 34 + ((e >> 2) << 4) + (half << 3) + ((e & 3) << 1));
    }
#pragma unroll
    for (int mi = 0; mi < 4; ++mi)
#pragma unroll
      for (int ni = 0; ni < 2; ++ni)
        acc[mi][ni] = __builtin_amdgcn_wmma_f32_16x16x32_bf16(
            false, a[mi].v, false, b[ni].v, (short)0, acc[mi][ni], false, false);
    __syncthreads();
  }

  // ---- epilogue ----
#pragma unroll
  for (int mi = 0; mi < 4; ++mi) {
#pragma unroll
    for (int ni = 0; ni < 2; ++ni) {
      int n = n0 + wn * 32 + ni * 16 + l16;
      if (n >= Nn) continue;
      float bsv = bias ? bias[n] : 0.0f;
#pragma unroll
      for (int r = 0; r < 8; ++r) {
        int m = m0 + wm * 64 + mi * 16 + half * 8 + r;
        float v = acc[mi][ni][r] + bsv;
        if (RELU) v = v > 0.0f ? v : 0.0f;
        if (OUT_BF16) Cb[(size_t)m * Nn + n] = (bf16_t)v;
        else          Cf[(size_t)m * Nn + n] = v;
      }
    }
  }
}

// ---------------- value transpose + padding mask ----------------

__global__ void k_value_transpose(const float* __restrict__ vin,
                                  const unsigned char* __restrict__ kpm,
                                  float* __restrict__ vout) {
  long t = (long)blockIdx.x * 256 + threadIdx.x;   // < MROWS*DM_
  int n = (int)(t & (DM_ - 1));
  int m = (int)(t >> 10);
  int b = m & (BZ_ - 1);
  int l = m >> 3;
  int h = n >> 6;
  int c = n & (HD_ - 1);
  float v = vin[t];
  if (kpm[b * LQ_ + l]) v = 0.0f;
  vout[((size_t)(b * NH_ + h) * LQ_ + l) * HD_ + c] = v;
}

// ---------------- fused softmax(K=4) + linear grid-sample + weighted sum ----------------

__global__ void k_sample(const float* __restrict__ value_t,
                         const float* __restrict__ offs,
                         const float* __restrict__ attnl,
                         const float* __restrict__ refp,
                         const float* __restrict__ snip,
                         bf16_t* __restrict__ outp) {
  long t = (long)blockIdx.x * 256 + threadIdx.x;   // < MROWS*NH_*HD_
  int c = (int)(t & (HD_ - 1));
  int h = (int)((t >> 6) & (NH_ - 1));
  int m = (int)(t >> 10);
  int b = m & (BZ_ - 1);
  int l = m >> 3;

  const float* al = attnl + (size_t)m * (NH_ * NK_) + h * NK_;
  const float* of = offs  + (size_t)m * (NH_ * NK_) + h * NK_;
  float a0 = al[0], a1 = al[1], a2 = al[2], a3 = al[3];
  float mx = fmaxf(fmaxf(a0, a1), fmaxf(a2, a3));
  float e0 = expf(a0 - mx), e1 = expf(a1 - mx), e2 = expf(a2 - mx), e3 = expf(a3 - mx);
  float inv = 1.0f / (e0 + e1 + e2 + e3);
  float w[4] = { e0 * inv, e1 * inv, e2 * inv, e3 * inv };

  float ref = refp[b * LQ_ + l];
  float sn  = snip[b];
  const float* vrow = value_t + (size_t)(b * NH_ + h) * LQ_ * HD_;

  float acc = 0.0f;
#pragma unroll
  for (int k = 0; k < NK_; ++k) {
    float xpos = (ref + of[k] / sn) * (float)(LQ_ - 1);
    float x0f = floorf(xpos);
    int x0 = (int)x0f;
    int x1 = x0 + 1;
    float w1 = xpos - x0f;
    float w0 = 1.0f - w1;
    float v0 = (x0 >= 0 && x0 < LQ_) ? vrow[(size_t)x0 * HD_ + c] : 0.0f;
    float v1 = (x1 >= 0 && x1 < LQ_) ? vrow[(size_t)x1 * HD_ + c] : 0.0f;
    acc += w[k] * (w0 * v0 + w1 * v1);
  }
  outp[(size_t)m * DM_ + h * HD_ + c] = (bf16_t)acc;
}

// ---------------- LayerNorm(residual a + r), wave32 per row ----------------

__global__ __launch_bounds__(256)
void k_ln_residual(const float* __restrict__ a, const float* __restrict__ r,
                   const float* __restrict__ g, const float* __restrict__ be,
                   float* __restrict__ outf, bf16_t* __restrict__ outb) {
  const int lane = threadIdx.x & 31;
  const int wid  = threadIdx.x >> 5;
  const size_t row = (size_t)blockIdx.x * 8 + wid;
  const float* pa = a + row * DM_;
  const float* pr = r + row * DM_;

  float vals[DM_ / 32];
  float s = 0.0f, s2 = 0.0f;
#pragma unroll
  for (int i = 0; i < DM_ / 32; ++i) {
    float v = pa[lane + 32 * i] + pr[lane + 32 * i];
    vals[i] = v;
    s += v;
    s2 += v * v;
  }
#pragma unroll
  for (int o = 16; o > 0; o >>= 1) {
    s  += __shfl_xor(s,  o, 32);
    s2 += __shfl_xor(s2, o, 32);
  }
  float mean = s * (1.0f / DM_);
  float var  = s2 * (1.0f / DM_) - mean * mean;
  float rstd = rsqrtf(var + 1e-5f);
#pragma unroll
  for (int i = 0; i < DM_ / 32; ++i) {
    int col = lane + 32 * i;
    float o = (vals[i] - mean) * rstd * g[col] + be[col];
    outf[row * DM_ + col] = o;
    if (outb) outb[row * DM_ + col] = (bf16_t)o;
  }
}

// ---------------- launch ----------------

extern "C" void kernel_launch(void* const* d_in, const int* in_sizes, int n_in,
                              void* d_out, int out_size, void* d_ws, size_t ws_size,
                              hipStream_t stream) {
  const float* src  = (const float*)d_in[0];
  const float* pos  = (const float*)d_in[1];
  const unsigned char* kpm = (const unsigned char*)d_in[2];
  const float* refp = (const float*)d_in[3];
  const float* snip = (const float*)d_in[4];
  const float* Wv   = (const float*)d_in[5];
  const float* bv   = (const float*)d_in[6];
  const float* Woff = (const float*)d_in[7];
  const float* boff = (const float*)d_in[8];
  const float* Wa   = (const float*)d_in[9];
  const float* ba   = (const float*)d_in[10];
  const float* Wo   = (const float*)d_in[11];
  const float* bo   = (const float*)d_in[12];
  const float* W1   = (const float*)d_in[13];
  const float* b1   = (const float*)d_in[14];
  const float* W2   = (const float*)d_in[15];
  const float* b2   = (const float*)d_in[16];
  const float* g1   = (const float*)d_in[17];
  const float* be1  = (const float*)d_in[18];
  const float* g2   = (const float*)d_in[19];
  const float* be2  = (const float*)d_in[20];
  float* out = (float*)d_out;

  char* ws = (char*)d_ws;
  size_t off = 0;
  auto alloc = [&](size_t bytes) -> void* {
    void* p = ws + off;
    off = (off + bytes + 255) & ~(size_t)255;
    return p;
  };

  // persistent bf16 weights, pre-transposed to [N][K]
  bf16_t* wWv   = (bf16_t*)alloc((size_t)DM_ * DM_ * 2);
  bf16_t* wWoff = (bf16_t*)alloc((size_t)DM_ * NH_ * NK_ * 2);
  bf16_t* wWa   = (bf16_t*)alloc((size_t)DM_ * NH_ * NK_ * 2);
  bf16_t* wWo   = (bf16_t*)alloc((size_t)DM_ * DM_ * 2);
  bf16_t* wW1   = (bf16_t*)alloc((size_t)DM_ * FF_ * 2);
  bf16_t* wW2   = (bf16_t*)alloc((size_t)FF_ * DM_ * 2);
  // activations (aliased)
  bf16_t* q_bf   = (bf16_t*)alloc((size_t)MROWS * DM_ * 2);
  bf16_t* src_bf = (bf16_t*)alloc((size_t)MROWS * DM_ * 2);
  float*  bufA   = (float*) alloc((size_t)MROWS * DM_ * 4);
  float*  value_t= (float*) alloc((size_t)BZ_ * NH_ * LQ_ * HD_ * 4);
  float*  offs   = (float*) alloc((size_t)MROWS * NH_ * NK_ * 4);
  float*  attnl  = (float*) alloc((size_t)MROWS * NH_ * NK_ * 4);
  bf16_t* attn_pre = (bf16_t*)alloc((size_t)MROWS * DM_ * 2);
  bf16_t* h_bf  = q_bf;        // q_bf + src_bf region reused for FFN hidden
  float*  x_f32 = value_t;     // reused after sampling
  bf16_t* x_bf  = attn_pre;    // reused after Wo GEMM

  const long nEl = (long)MROWS * DM_;

  // 1. convert + transpose weights to bf16 [N][K]
  auto cvt = [&](const float* in, bf16_t* o, int K, int N) {
    long n = (long)K * N;
    k_cvt_tr_bf16<<<(unsigned)((n + 255) / 256), 256, 0, stream>>>(in, o, K, N);
  };
  cvt(Wv,   wWv,   DM_, DM_);
  cvt(Woff, wWoff, DM_, NH_ * NK_);
  cvt(Wa,   wWa,   DM_, NH_ * NK_);
  cvt(Wo,   wWo,   DM_, DM_);
  cvt(W1,   wW1,   DM_, FF_);
  cvt(W2,   wW2,   FF_, DM_);

  // 2. q = src + pos (bf16), src -> bf16
  k_prep<<<(unsigned)((nEl + 255) / 256), 256, 0, stream>>>(src, pos, q_bf, src_bf, nEl);

  dim3 blk(256);
  dim3 g1024(DM_ / 128, MROWS / 128);
  dim3 g64(1, MROWS / 128);
  dim3 g2048(FF_ / 128, MROWS / 128);

  // 3. value = src @ Wv + bv
  k_gemm_bf16<false, false><<<g1024, blk, 0, stream>>>(src_bf, wWv, bv, bufA, nullptr,
                                                       MROWS, DM_, DM_);
  // 4. transpose + mask into [bz*H][Lv][HD]
  k_value_transpose<<<(unsigned)(nEl / 256), 256, 0, stream>>>(bufA, kpm, value_t);

  // 5/6. offset and attention projections
  k_gemm_bf16<false, false><<<g64, blk, 0, stream>>>(q_bf, wWoff, boff, offs, nullptr,
                                                     MROWS, NH_ * NK_, DM_);
  k_gemm_bf16<false, false><<<g64, blk, 0, stream>>>(q_bf, wWa, ba, attnl, nullptr,
                                                     MROWS, NH_ * NK_, DM_);

  // 7. softmax + grid-sample + weighted sum
  k_sample<<<(unsigned)(((long)MROWS * NH_ * HD_) / 256), 256, 0, stream>>>(
      value_t, offs, attnl, refp, snip, attn_pre);

  // 8. attn_out = attn_pre @ Wo + bo
  k_gemm_bf16<false, false><<<g1024, blk, 0, stream>>>(attn_pre, wWo, bo, bufA, nullptr,
                                                       MROWS, DM_, DM_);
  // 9. x = LN(src + attn_out)
  k_ln_residual<<<MROWS / 8, 256, 0, stream>>>(src, bufA, g1, be1, x_f32, x_bf);

  // 10. h = relu(x @ W1 + b1)  (bf16 out)
  k_gemm_bf16<true, true><<<g2048, blk, 0, stream>>>(x_bf, wW1, b1, nullptr, h_bf,
                                                     MROWS, FF_, DM_);
  // 11. y = h @ W2 + b2
  k_gemm_bf16<false, false><<<g1024, blk, 0, stream>>>(h_bf, wW2, b2, bufA, nullptr,
                                                       MROWS, DM_, FF_);
  // 12. out = LN(x + y)
  k_ln_residual<<<MROWS / 8, 256, 0, stream>>>(x_f32, bufA, g2, be2, out, nullptr);
}
